// NetGAT_6777458393522
// MI455X (gfx1250) — matrix operand order repaired and checked by reading, not compiled, not run on hardware
//
#include <hip/hip_runtime.h>
#include <hip/hip_bf16.h>
#include <math.h>

typedef __attribute__((ext_vector_type(16))) _Float16 v16h;
typedef __attribute__((ext_vector_type(8)))  float    v8f;

#define NODES 50000
#define EDGES 800000

// ---------------------------------------------------------------------------
// Ordered-int float atomic max (works for mixed signs with -inf init).
// ---------------------------------------------------------------------------
__device__ __forceinline__ void atomic_max_float(float* addr, float v) {
  if (v >= 0.0f) atomicMax((int*)addr, __float_as_int(v));
  else           atomicMin((unsigned int*)addr, __float_as_uint(v));
}

// ---------------------------------------------------------------------------
// Pre-pack a weight matrix W[K,Nout] (f32, row-major) into the WMMA B-fragment
// lane layout (f16), zero-padded so K rounds up to a multiple of 32.
// B layout (16-bit, 32x16): j 0..7 -> K = 8*hi + j ; j 8..15 -> K = 16 + 8*hi + (j-8), N = lane&15.
// ---------------------------------------------------------------------------
__global__ void pack_w(const float* __restrict__ W, _Float16* __restrict__ Wp,
                       int K, int Nout, int total) {
  int gid = blockIdx.x * blockDim.x + threadIdx.x;
  if (gid >= total) return;
  int j    = gid & 15;
  int lane = (gid >> 4) & 31;
  int tt   = gid >> 9;                 // tt = s * tiles_n + t
  int tiles_n = Nout >> 4;
  int s = tt / tiles_n;
  int t = tt - s * tiles_n;
  int hi = lane >> 4, l15 = lane & 15;
  int kb = s * 32 + (j < 8 ? 8 * hi + j : 16 + 8 * hi + (j - 8));
  int n  = t * 16 + l15;
  float v = (kb < K) ? W[(size_t)kb * Nout + n] : 0.0f;
  Wp[gid] = (_Float16)v;
}

// ---------------------------------------------------------------------------
// Load one A fragment (16-bit 16x32 layout) from an f32 row, unguarded.
// j 0..7 -> K = kk + 8*hi + j ; j 8..15 -> K = kk + 16 + 8*hi + (j-8)
// ---------------------------------------------------------------------------
__device__ __forceinline__ v16h load_a_frag(const float* __restrict__ xr,
                                            int kk, int hi) {
  float4 f0 = *(const float4*)(xr + kk + 8 * hi);
  float4 f1 = *(const float4*)(xr + kk + 8 * hi + 4);
  float4 f2 = *(const float4*)(xr + kk + 16 + 8 * hi);
  float4 f3 = *(const float4*)(xr + kk + 16 + 8 * hi + 4);
  v16h a;
  a[0]  = (_Float16)f0.x; a[1]  = (_Float16)f0.y; a[2]  = (_Float16)f0.z; a[3]  = (_Float16)f0.w;
  a[4]  = (_Float16)f1.x; a[5]  = (_Float16)f1.y; a[6]  = (_Float16)f1.z; a[7]  = (_Float16)f1.w;
  a[8]  = (_Float16)f2.x; a[9]  = (_Float16)f2.y; a[10] = (_Float16)f2.z; a[11] = (_Float16)f2.w;
  a[12] = (_Float16)f3.x; a[13] = (_Float16)f3.y; a[14] = (_Float16)f3.z; a[15] = (_Float16)f3.w;
  return a;
}

// ---------------------------------------------------------------------------
// Fused Q/K/V/skip projections: 4 outputs sharing one A fragment per tile.
// Wp holds 4 packed weight slots of 4096 halfs each (q,k,v,s).
// K is 32 or 64 (layer inputs) -> no padding paths.
// ---------------------------------------------------------------------------
template <int K>
__global__ void gemm_qkvs(const float* __restrict__ X,
                          const _Float16* __restrict__ Wp,
                          const float* __restrict__ bq, const float* __restrict__ bk,
                          const float* __restrict__ bv, const float* __restrict__ bs,
                          float* __restrict__ yq, float* __restrict__ yk,
                          float* __restrict__ yv, float* __restrict__ ys,
                          int M, int Nout) {
  const int wave = (int)((blockIdx.x * blockDim.x + threadIdx.x) >> 5);
  const int lane = threadIdx.x & 31;
  const int tiles_n = Nout >> 4;
  const int tiles_m = M >> 4;
  if (wave >= tiles_m * tiles_n) return;       // wave-uniform: EXEC all-1s at WMMA
  const int tm  = wave / tiles_n;
  const int tn  = wave - tm * tiles_n;
  const int l15 = lane & 15;
  const int hi  = lane >> 4;
  const float* xr = X + (size_t)(tm * 16 + l15) * K;
  const int colB  = tn * 16 + l15;
  const v16h* wv  = (const v16h*)Wp;           // 256 v16h per slot

  constexpr int KS = K / 32;
  v16h a[KS];
#pragma unroll
  for (int s = 0; s < KS; ++s) a[s] = load_a_frag(xr, s * 32, hi);

  v8f acc[4];
#pragma unroll
  for (int i = 0; i < 4; ++i) {
    v8f c = {};
#pragma unroll
    for (int s = 0; s < KS; ++s) {
      v16h b = wv[(size_t)i * 256 + (size_t)(s * tiles_n + tn) * 32 + lane];
      c = __builtin_amdgcn_wmma_f32_16x16x32_f16(false, a[s], false, b,
                                                 (short)0, c, false, false);
    }
    acc[i] = c;
  }

  const float* biases[4] = {bq, bk, bv, bs};
  float*       outs[4]   = {yq, yk, yv, ys};
#pragma unroll
  for (int i = 0; i < 4; ++i) {
    float bn = biases[i][colB];
#pragma unroll
    for (int r = 0; r < 8; ++r) {
      int row = tm * 16 + r + 8 * hi;          // C/D: VGPR r -> M = r + 8*hi
      outs[i][(size_t)row * Nout + colB] = acc[i][r] + bn;
    }
  }
}

// ---------------------------------------------------------------------------
// Single-output GEMM for the readout MLP (K may be 48 -> compile-time padding).
// act: 0 = identity, 1 = tanhshrink (z - tanh(z)).
// ---------------------------------------------------------------------------
template <int K>
__global__ void gemm_bias_act(const float* __restrict__ X,
                              const _Float16* __restrict__ Wp,
                              const float* __restrict__ bias,
                              float* __restrict__ Y,
                              int M, int Nout, int act) {
  const int wave = (int)((blockIdx.x * blockDim.x + threadIdx.x) >> 5);
  const int lane = threadIdx.x & 31;
  const int tiles_n = Nout >> 4;
  const int tiles_m = M >> 4;
  if (wave >= tiles_m * tiles_n) return;
  const int tm  = wave / tiles_n;
  const int tn  = wave - tm * tiles_n;
  const int l15 = lane & 15;
  const int hi  = lane >> 4;
  const float* xr = X + (size_t)(tm * 16 + l15) * K;
  const int colB  = tn * 16 + l15;
  const v16h* wv  = (const v16h*)Wp;

  constexpr int KS = (K + 31) / 32;
  v8f acc = {};
#pragma unroll
  for (int s = 0; s < KS; ++s) {
    const int kk = s * 32;
    v16h a;
    if (kk + 16 <= K) {                        // compile-time after unroll
      float4 f0 = *(const float4*)(xr + kk + 8 * hi);
      float4 f1 = *(const float4*)(xr + kk + 8 * hi + 4);
      a[0] = (_Float16)f0.x; a[1] = (_Float16)f0.y; a[2] = (_Float16)f0.z; a[3] = (_Float16)f0.w;
      a[4] = (_Float16)f1.x; a[5] = (_Float16)f1.y; a[6] = (_Float16)f1.z; a[7] = (_Float16)f1.w;
    } else {
#pragma unroll
      for (int j = 0; j < 8; ++j) a[j] = (_Float16)0.0f;
    }
    if (kk + 32 <= K) {
      float4 f2 = *(const float4*)(xr + kk + 16 + 8 * hi);
      float4 f3 = *(const float4*)(xr + kk + 16 + 8 * hi + 4);
      a[8]  = (_Float16)f2.x; a[9]  = (_Float16)f2.y; a[10] = (_Float16)f2.z; a[11] = (_Float16)f2.w;
      a[12] = (_Float16)f3.x; a[13] = (_Float16)f3.y; a[14] = (_Float16)f3.z; a[15] = (_Float16)f3.w;
    } else {
#pragma unroll
      for (int j = 8; j < 16; ++j) a[j] = (_Float16)0.0f;
    }
    v16h b = wv[(size_t)(s * tiles_n + tn) * 32 + lane];
    acc = __builtin_amdgcn_wmma_f32_16x16x32_f16(false, a, false, b,
                                                 (short)0, acc, false, false);
  }

  const float bn = bias[colB];
#pragma unroll
  for (int r = 0; r < 8; ++r) {
    int row = tm * 16 + r + 8 * hi;
    float v = acc[r] + bn;
    if (act == 1) v = v - tanhf(v);
    Y[(size_t)row * Nout + colB] = v;
  }
}

// ---------------------------------------------------------------------------
// segment-softmax scratch init
// ---------------------------------------------------------------------------
__global__ void seg_init(float* __restrict__ segmax, float* __restrict__ segsum, int n) {
  int i = blockIdx.x * blockDim.x + threadIdx.x;
  if (i < n) { segmax[i] = -__builtin_inff(); segsum[i] = 0.0f; }
}

// ---------------------------------------------------------------------------
// Pass A: per-(edge,head) attention logit + segment max (float4 gathers).
// ---------------------------------------------------------------------------
__global__ void edge_logits_max(const float* __restrict__ q,
                                const float* __restrict__ k,
                                const int* __restrict__ src,
                                const int* __restrict__ dst,
                                float* __restrict__ logits,
                                float* __restrict__ segmax,
                                int H, int C, float scale) {
  int gid = blockIdx.x * blockDim.x + threadIdx.x;
  if (gid >= EDGES * H) return;
  int e = gid / H, h = gid - e * H;
  int s = src[e], d = dst[e];
  const float4* qr = (const float4*)(q + (size_t)d * (H * C) + (size_t)h * C);
  const float4* kr = (const float4*)(k + (size_t)s * (H * C) + (size_t)h * C);
  float dot = 0.0f;
  for (int c4 = 0; c4 < (C >> 2); ++c4) {
    float4 a = qr[c4], b = kr[c4];
    dot += a.x * b.x + a.y * b.y + a.z * b.z + a.w * b.w;
  }
  dot *= scale;
  logits[gid] = dot;
  atomic_max_float(&segmax[(size_t)d * H + h], dot);
}

// ---------------------------------------------------------------------------
// Pass B: w = exp(logit - max[dst]); segment sum.  Overwrites logits with w.
// ---------------------------------------------------------------------------
__global__ void edge_exp_sum(const int* __restrict__ dst,
                             float* __restrict__ logits,
                             const float* __restrict__ segmax,
                             float* __restrict__ segsum, int H) {
  int gid = blockIdx.x * blockDim.x + threadIdx.x;
  if (gid >= EDGES * H) return;
  int e = gid / H, h = gid - e * H;
  int d = dst[e];
  float w = expf(logits[gid] - segmax[(size_t)d * H + h]);
  logits[gid] = w;
  atomicAdd(&segsum[(size_t)d * H + h], w);
}

// ---------------------------------------------------------------------------
// Pass C: out[dst] += alpha * v[src].  One thread per (edge, 4-column group);
// groups never straddle a head boundary (C is 32 or 64).
// ---------------------------------------------------------------------------
__global__ void edge_scatter(const float* __restrict__ v,
                             const int* __restrict__ src,
                             const int* __restrict__ dst,
                             const float* __restrict__ logits,
                             const float* __restrict__ segsum,
                             float* __restrict__ out, int H, int C) {
  int gid = blockIdx.x * blockDim.x + threadIdx.x;
  if (gid >= EDGES * 16) return;
  int e   = gid >> 4;
  int col = (gid & 15) << 2;
  int h   = col / C;
  int s = src[e], d = dst[e];
  float alpha = logits[(size_t)e * H + h] / segsum[(size_t)d * H + h];
  const float4 vv = *(const float4*)(v + (size_t)s * 64 + col);
  float* o = out + (size_t)d * 64 + col;
  atomicAdd(o + 0, alpha * vv.x);
  atomicAdd(o + 1, alpha * vv.y);
  atomicAdd(o + 2, alpha * vv.z);
  atomicAdd(o + 3, alpha * vv.w);
}

// ---------------------------------------------------------------------------
// Host driver
// ---------------------------------------------------------------------------
extern "C" void kernel_launch(void* const* d_in, const int* in_sizes, int n_in,
                              void* d_out, int out_size, void* d_ws, size_t ws_size,
                              hipStream_t stream) {
  (void)in_sizes; (void)n_in; (void)out_size; (void)ws_size;

  const float* x  = (const float*)d_in[0];
  const int*   ei = (const int*)d_in[1];
  const int* src = ei;
  const int* dst = ei + EDGES;

  // params flatten order: c1{Wq,bq,Wk,bk,Wv,bv,Ws,bs}, c2{...}, c3{...},
  //                       ro{W1,b1,W2,b2,W3,b3}
  const float* c1[8]; const float* c2[8]; const float* c3[8];
  for (int i = 0; i < 8; ++i) {
    c1[i] = (const float*)d_in[2 + i];
    c2[i] = (const float*)d_in[10 + i];
    c3[i] = (const float*)d_in[18 + i];
  }
  const float* W1 = (const float*)d_in[26]; const float* b1 = (const float*)d_in[27];
  const float* W2 = (const float*)d_in[28]; const float* b2 = (const float*)d_in[29];
  const float* W3 = (const float*)d_in[30]; const float* b3 = (const float*)d_in[31];

  // workspace carve-up (floats)
  float* ws = (float*)d_ws;
  float* hA     = ws;                         // [N,64]
  float* hB     = hA + (size_t)NODES * 64;    // [N,64]
  float* qbuf   = hB + (size_t)NODES * 64;    // [N,64]  (reused as z1 [N,48])
  float* kbuf   = qbuf + (size_t)NODES * 64;  // [N,64]  (reused as z2 [N,48])
  float* vbuf   = kbuf + (size_t)NODES * 64;  // [N,64]
  float* segmax = vbuf + (size_t)NODES * 64;  // [N,2]
  float* segsum = segmax + (size_t)NODES * 2; // [N,2]
  float* logits = segsum + (size_t)NODES * 2; // [E,2]
  _Float16* wpack = (_Float16*)(logits + (size_t)EDGES * 2); // 15 x 4096 halfs

  auto wslot = [&](int i) { return wpack + (size_t)i * 4096; };

  auto pack = [&](const float* W, int K, int Nout, _Float16* dstp) {
    int KS = (K + 31) / 32;
    int total = KS * (Nout >> 4) * 512;
    pack_w<<<(total + 255) / 256, 256, 0, stream>>>(W, dstp, K, Nout, total);
  };

  // pack all weights up front: slots 0..3 c1, 4..7 c2, 8..11 c3, 12..14 readout
  for (int i = 0; i < 4; ++i) {
    pack(c1[2 * i], 32, 64, wslot(i));
    pack(c2[2 * i], 64, 64, wslot(4 + i));
    pack(c3[2 * i], 64, 64, wslot(8 + i));
  }
  pack(W1, 64, 48, wslot(12));
  pack(W2, 48, 48, wslot(13));
  pack(W3, 48, 16, wslot(14));

  auto gemm_ro = [&](const float* X, const _Float16* Wp, const float* B, float* Y,
                     int K, int Nout, int act) {
    int tiles = (NODES >> 4) * (Nout >> 4);
    int blocks = (tiles + 7) / 8;                      // 8 waves / 256 threads
    switch (K) {
      case 48: gemm_bias_act<48><<<blocks, 256, 0, stream>>>(X, Wp, B, Y, NODES, Nout, act); break;
      default: gemm_bias_act<64><<<blocks, 256, 0, stream>>>(X, Wp, B, Y, NODES, Nout, act); break;
    }
  };

  auto layer = [&](const float* hin, int K, int H, int C,
                   const float* const* p, const _Float16* wp0, float* hout) {
    int tiles = (NODES >> 4) * 4;                      // Nout = 64 -> 4 n-tiles
    int blocks = (tiles + 7) / 8;
    if (K == 32)
      gemm_qkvs<32><<<blocks, 256, 0, stream>>>(hin, wp0, p[1], p[3], p[5], p[7],
                                                qbuf, kbuf, vbuf, hout, NODES, 64);
    else
      gemm_qkvs<64><<<blocks, 256, 0, stream>>>(hin, wp0, p[1], p[3], p[5], p[7],
                                                qbuf, kbuf, vbuf, hout, NODES, 64);
    int nh = NODES * H;
    seg_init<<<(nh + 255) / 256, 256, 0, stream>>>(segmax, segsum, nh);
    float scale = 1.0f / sqrtf((float)C);
    int eh = EDGES * H;
    edge_logits_max<<<(eh + 255) / 256, 256, 0, stream>>>(qbuf, kbuf, src, dst,
                                                          logits, segmax, H, C, scale);
    edge_exp_sum<<<(eh + 255) / 256, 256, 0, stream>>>(dst, logits, segmax, segsum, H);
    int es = EDGES * 16;
    edge_scatter<<<(es + 255) / 256, 256, 0, stream>>>(vbuf, src, dst, logits,
                                                       segsum, hout, H, C);
  };

  layer(x,  32, 2, 32, c1, wslot(0), hA);   // conv1: heads=2, ch=32, concat -> 64
  layer(hA, 64, 1, 64, c2, wslot(4), hB);   // conv2
  layer(hB, 64, 1, 64, c3, wslot(8), hA);   // conv3

  // readout MLP with fused tanhshrink epilogues
  gemm_ro(hA,   wslot(12), b1, qbuf, 64, 48, 1);
  gemm_ro(qbuf, wslot(13), b2, kbuf, 48, 48, 1);
  gemm_ro(kbuf, wslot(14), b3, (float*)d_out, 48, 16, 0);
}
